// GroupedQueryAttention_40827959116477
// MI455X (gfx1250) — matrix operand order repaired and checked by reading, not compiled, not run on hardware
//
#include <hip/hip_runtime.h>

// ---------------------------------------------------------------------------
// GQA forward for MI455X (gfx1250, wave32, WMMA).
//   out  = softmax(mask(QK^T * scale)) @ V  merged @ Wo     (fp32)
//   attn = the softmax probabilities                        (fp32, 1 GiB)
// All GEMMs on v_wmma_f32_16x16x32_bf16, fp32 accumulation.
// Block tile 128x128, 8 wave32s, wave tile 32x64 (2x4 WMMA frags), BK=32.
// LDS double-buffered: A row-major [2][128][40], B fragment-major [2][128][40]
// -> every WMMA operand is two ds_load_b128, one barrier per K-step.
// Pure-bf16 row-major tiles stream via global_load_async_to_lds_b128
// (ASYNCcnt); fp32 tiles stage through registers (loads overlap the WMMAs).
// ---------------------------------------------------------------------------

typedef __attribute__((ext_vector_type(16))) __bf16 v16bf;
typedef __attribute__((ext_vector_type(8)))  __bf16 v8bf;
typedef __attribute__((ext_vector_type(4)))  __bf16 v4bf;
typedef __attribute__((ext_vector_type(8)))  float  v8f;
typedef __attribute__((ext_vector_type(4)))  int    v4i;

#define BATCH     2
#define SEQ       2048
#define DMODEL    4096
#define NHEADS    32
#define NKVHEADS  8
#define HEADDIM   128
#define KVGROUPS  4
#define KVDIM     (NKVHEADS * HEADDIM)   // 1024
#define ATTNSCALE 0.08838834764831845f   // 128^-0.5
#define FP32_MIN  (-3.4028234663852886e+38f)

#define BM 128
#define BN 128
#define BK 32
#define LP 40   // padded LDS inner stride (halves): 80 B, 16 B aligned

#if __has_builtin(__builtin_amdgcn_global_load_async_to_lds_b128)
#define HAVE_ASYNC_LDS 1
#else
#define HAVE_ASYNC_LDS 0
#endif

__device__ __forceinline__ void wait_async0() {
#if __has_builtin(__builtin_amdgcn_s_wait_asynccnt)
    __builtin_amdgcn_s_wait_asynccnt(0);
#else
    asm volatile("s_wait_asynccnt 0x0" ::: "memory");
#endif
}

__device__ __forceinline__ void st_c(float* p, float v)  { *p = v; }
__device__ __forceinline__ void st_c(__bf16* p, float v) { *p = (__bf16)v; }

template <typename T> struct IsBF16          { static constexpr bool value = false; };
template <>           struct IsBF16<__bf16>  { static constexpr bool value = true;  };

// ---- per-thread register block holding one 128x32 (or 32x128) tile slice ---
template <typename T> struct Regs;
template <> struct Regs<float>  { float4 v[4]; };   // 16 floats
template <> struct Regs<__bf16> { v8bf  v[2]; };    // 16 halves

// ---- row-major copy staging (A tiles, and B^T tiles): src pre-offset ------
__device__ __forceinline__ void load_rm(Regs<float>& r, const float* src,
                                        int ld, int tid) {
#pragma unroll
    for (int it = 0; it < 4; ++it) {
        const int q = tid + it * 256;          // 1024 quads: 128 rows x 8
        const int rr = q >> 3, c4 = (q & 7) << 2;
        r.v[it] = *(const float4*)(src + (long long)rr * ld + c4);
    }
}
__device__ __forceinline__ void load_rm(Regs<__bf16>& r, const __bf16* src,
                                        int ld, int tid) {
#pragma unroll
    for (int it = 0; it < 2; ++it) {
        const int o = tid + it * 256;          // 512 octets: 128 rows x 4
        const int rr = o >> 2, c8 = (o & 3) << 3;
        r.v[it] = *(const v8bf*)(src + (long long)rr * ld + c8);
    }
}
__device__ __forceinline__ void store_rm(__bf16 (*Ls)[LP], const Regs<float>& r,
                                         int tid) {
#pragma unroll
    for (int it = 0; it < 4; ++it) {
        const int q = tid + it * 256;
        const int rr = q >> 3, c4 = (q & 7) << 2;
        const float4 f = r.v[it];
        v4bf b; b[0] = (__bf16)f.x; b[1] = (__bf16)f.y;
                b[2] = (__bf16)f.z; b[3] = (__bf16)f.w;
        *(v4bf*)&Ls[rr][c4] = b;
    }
}
__device__ __forceinline__ void store_rm(__bf16 (*Ls)[LP], const Regs<__bf16>& r,
                                         int tid) {
#pragma unroll
    for (int it = 0; it < 2; ++it) {
        const int o = tid + it * 256;
        const int rr = o >> 2, c8 = (o & 3) << 3;
        *(v8bf*)&Ls[rr][c8] = r.v[it];
    }
}

// ---- async global->LDS row-major bf16 copy (no VGPR round trip) -----------
#if HAVE_ASYNC_LDS
__device__ __forceinline__ void async_rm(__bf16 (*Ls)[LP], const __bf16* src,
                                         int ld, int tid) {
#pragma unroll
    for (int it = 0; it < 2; ++it) {
        const int o = tid + it * 256;          // 512 octets: 128 rows x 4
        const int rr = o >> 2, c8 = (o & 3) << 3;
        __builtin_amdgcn_global_load_async_to_lds_b128(
            (__attribute__((address_space(1))) v4i*)(src + (long long)rr * ld + c8),
            (__attribute__((address_space(3))) v4i*)&Ls[rr][c8],
            0, 0);
    }
}
#else
__device__ __forceinline__ void async_rm(__bf16 (*)[LP], const __bf16*, int, int) {}
#endif

// ---- non-transposed B (stored [K,N], pre-offset to k0*ldb+colBase): --------
// coalesced loads along N, transposed scatter into Bt[n][k].
__device__ __forceinline__ void load_nt(Regs<float>& r, const float* src,
                                        int ldb, int tid) {
#pragma unroll
    for (int it = 0; it < 4; ++it) {
        const int q = tid + it * 256;          // 1024 quads: 32 k x 32
        const int kk = q >> 5, n4 = (q & 31) << 2;
        r.v[it] = *(const float4*)(src + (long long)kk * ldb + n4);
    }
}
__device__ __forceinline__ void load_nt(Regs<__bf16>& r, const __bf16* src,
                                        int ldb, int tid) {
#pragma unroll
    for (int it = 0; it < 2; ++it) {
        const int o = tid + it * 256;          // 512 octets: 32 k x 16
        const int kk = o >> 4, n8 = (o & 15) << 3;
        r.v[it] = *(const v8bf*)(src + (long long)kk * ldb + n8);
    }
}
__device__ __forceinline__ void store_nt(__bf16 (*Bt)[LP], const Regs<float>& r,
                                         int tid) {
#pragma unroll
    for (int it = 0; it < 4; ++it) {
        const int q = tid + it * 256;
        const int kk = q >> 5, n4 = (q & 31) << 2;
        const float4 f = r.v[it];
        Bt[n4 + 0][kk] = (__bf16)f.x; Bt[n4 + 1][kk] = (__bf16)f.y;
        Bt[n4 + 2][kk] = (__bf16)f.z; Bt[n4 + 3][kk] = (__bf16)f.w;
    }
}
__device__ __forceinline__ void store_nt(__bf16 (*Bt)[LP], const Regs<__bf16>& r,
                                         int tid) {
#pragma unroll
    for (int it = 0; it < 2; ++it) {
        const int o = tid + it * 256;
        const int kk = o >> 4, n8 = (o & 15) << 3;
#pragma unroll
        for (int j = 0; j < 8; ++j) Bt[n8 + j][kk] = r.v[it][j];
    }
}

__device__ __forceinline__ v16bf frag16(const __bf16* p0, const __bf16* p1) {
    const v8bf lo = *(const v8bf*)p0;
    const v8bf hi = *(const v8bf*)p1;
    return __builtin_shufflevector(lo, hi, 0, 1, 2, 3, 4, 5, 6, 7,
                                           8, 9, 10, 11, 12, 13, 14, 15);
}

// ---------------------------------------------------------------------------
template <typename TA, typename TB, typename TC, bool TRANSB, bool MASK, bool CAUSAL_A>
__global__ __launch_bounds__(256) void gemm_wmma_kernel(
    const TA* __restrict__ Ag, const TB* __restrict__ Bg, TC* __restrict__ Cg,
    int M, int N, int K, int lda, int ldb, int ldc,
    int Hq, int group,
    long long sAb, long long sAh, long long sBb, long long sBh,
    long long sCb, long long sCh,
    float alpha, const int* __restrict__ maskPtr)
{
    __shared__ __bf16 As[2][BM][LP];   // A row-major:      As[buf][m][k]
    __shared__ __bf16 Bt[2][BN][LP];   // B fragment-major: Bt[buf][n][k]

    constexpr bool ASYNC_A = HAVE_ASYNC_LDS && IsBF16<TA>::value;            // A: row-major copy
    constexpr bool ASYNC_B = HAVE_ASYNC_LDS && TRANSB && IsBF16<TB>::value;  // B^T: row-major copy

    const int tid  = threadIdx.x;
    const int lane = tid & 31;
    const int wave = tid >> 5;
    const int wm   = (wave >> 1) * 32;   // 4 wave rows
    const int wn   = (wave & 1) * 64;    // 2 wave cols, each 64 wide

    const int z   = blockIdx.z;
    const int bz  = z / Hq;
    const int hz  = z % Hq;
    const int kvh = hz / group;

    const TA* A = Ag + (long long)bz * sAb + (long long)hz * sAh;
    const TB* B = Bg + (long long)bz * sBb + (long long)kvh * sBh;
    TC*       C = Cg + (long long)bz * sCb + (long long)hz * sCh;

    const int rowBase = blockIdx.y * BM;
    const int colBase = blockIdx.x * BN;

    v8f acc[2][4] = {};

    // Causal classification of this 128x128 tile.
    const bool fullMask   = MASK && (colBase >= rowBase + BM);           // all masked
    const bool needMaskRd = MASK && !fullMask && (colBase + BN - 1 > rowBase); // diagonal

    if (!fullMask) {
        int kEnd = K;
        if (CAUSAL_A) { const int lim = rowBase + BM; kEnd = lim < K ? lim : K; }

        const int akb   = (lane < 16) ? 0 : 8;    // A: K halves per lane group
        const int bkb   = (lane < 16) ? 0 : 8;    // B: K halves per lane group
        const int arow0 = wm + (lane & 15);
        const int bcol0 = wn + (lane & 15);

        const TA* Arow = A + (long long)rowBase * lda;
        const TB* Brow = TRANSB ? (B + (long long)colBase * ldb) : (B + colBase);

        Regs<TA> ra;
        Regs<TB> rb;

        // --- prologue: stage tile 0 into buffer 0
        if (ASYNC_A) async_rm((__bf16(*)[LP])As[0], (const __bf16*)Arow, lda, tid);
        else { load_rm(ra, Arow, lda, tid); store_rm(As[0], ra, tid); }
        if (TRANSB) {
            if (ASYNC_B) async_rm((__bf16(*)[LP])Bt[0], (const __bf16*)Brow, ldb, tid);
            else { load_rm(rb, Brow, ldb, tid); store_rm(Bt[0], rb, tid); }
        } else {
            load_nt(rb, Brow, ldb, tid); store_nt(Bt[0], rb, tid);
        }
        if (ASYNC_A || ASYNC_B) wait_async0();
        __syncthreads();

        int buf = 0;
        for (int k0 = 0; k0 < kEnd; k0 += BK, buf ^= 1) {
            const int  kn   = k0 + BK;
            const bool more = kn < kEnd;
            const int  nbuf = buf ^ 1;

            // --- issue next tile's global traffic; completes under the WMMAs
            if (more) {
                if (ASYNC_A) async_rm((__bf16(*)[LP])As[nbuf],
                                      (const __bf16*)(Arow + kn), lda, tid);
                else         load_rm(ra, Arow + kn, lda, tid);
                if (TRANSB) {
                    if (ASYNC_B) async_rm((__bf16(*)[LP])Bt[nbuf],
                                          (const __bf16*)(Brow + kn), ldb, tid);
                    else         load_rm(rb, Brow + kn, ldb, tid);
                } else {
                    load_nt(rb, B + (long long)kn * ldb + colBase, ldb, tid);
                }
            }

            // --- 8 WMMAs from current buffer
            v16bf afrag[2], bfrag[4];
#pragma unroll
            for (int mi = 0; mi < 2; ++mi) {
                const int m = arow0 + mi * 16;
                afrag[mi] = frag16(&As[buf][m][akb], &As[buf][m][16 + akb]);
            }
#pragma unroll
            for (int ni = 0; ni < 4; ++ni) {
                const int n = bcol0 + ni * 16;
                bfrag[ni] = frag16(&Bt[buf][n][bkb], &Bt[buf][n][16 + bkb]);
            }
#pragma unroll
            for (int mi = 0; mi < 2; ++mi)
#pragma unroll
                for (int ni = 0; ni < 4; ++ni)
                    acc[mi][ni] = __builtin_amdgcn_wmma_f32_16x16x32_bf16(
                        false, afrag[mi], false, bfrag[ni],
                        (short)0, acc[mi][ni], false, false);

            // --- drain register-staged tiles into the next buffer
            if (more) {
                if (!ASYNC_A) store_rm(As[nbuf], ra, tid);
                if (TRANSB) { if (!ASYNC_B) store_rm(Bt[nbuf], rb, tid); }
                else        store_nt(Bt[nbuf], rb, tid);
                if (ASYNC_A || ASYNC_B) wait_async0();
            }
            __syncthreads();
        }
    }

    // ---- store C: acc element r -> M = r + 8*(lane>=16), N = lane&15
    const int crow0 = rowBase + wm + ((lane >> 4) << 3);
    const int ccol0 = colBase + wn + (lane & 15);
#pragma unroll
    for (int mi = 0; mi < 2; ++mi) {
#pragma unroll
        for (int ni = 0; ni < 4; ++ni) {
            const int col = ccol0 + ni * 16;
#pragma unroll
            for (int r = 0; r < 8; ++r) {
                const int row = crow0 + mi * 16 + r;
                float v = acc[mi][ni][r] * alpha;
                if (MASK) {
                    if (fullMask) v = FP32_MIN;
                    else if (needMaskRd &&
                             maskPtr[(long long)row * N + col] == 0) v = FP32_MIN;
                }
                st_c(&C[(long long)row * ldc + col], v);
            }
        }
    }
}

// ---------------------------------------------------------------------------
// In-place row softmax over attn (one 256-thread block per row of 2048),
// float4 I/O: each thread owns two 16 B chunks.
// ---------------------------------------------------------------------------
__global__ __launch_bounds__(256) void softmax_rows_kernel(float* __restrict__ attn)
{
    float* p = attn + (long long)blockIdx.x * SEQ;
    const int tid = threadIdx.x;

    float4 a = *(const float4*)&p[tid * 4];
    float4 b = *(const float4*)&p[tid * 4 + 1024];
    float m = fmaxf(fmaxf(fmaxf(a.x, a.y), fmaxf(a.z, a.w)),
                    fmaxf(fmaxf(b.x, b.y), fmaxf(b.z, b.w)));

    __shared__ float red[256];
    red[tid] = m; __syncthreads();
    for (int off = 128; off > 0; off >>= 1) {
        if (tid < off) red[tid] = fmaxf(red[tid], red[tid + off]);
        __syncthreads();
    }
    m = red[0]; __syncthreads();

    a.x = __expf(a.x - m); a.y = __expf(a.y - m);
    a.z = __expf(a.z - m); a.w = __expf(a.w - m);
    b.x = __expf(b.x - m); b.y = __expf(b.y - m);
    b.z = __expf(b.z - m); b.w = __expf(b.w - m);
    const float s = (a.x + a.y + a.z + a.w) + (b.x + b.y + b.z + b.w);

    red[tid] = s; __syncthreads();
    for (int off = 128; off > 0; off >>= 1) {
        if (tid < off) red[tid] += red[tid + off];
        __syncthreads();
    }
    const float inv = 1.0f / red[0];

    a.x *= inv; a.y *= inv; a.z *= inv; a.w *= inv;
    b.x *= inv; b.y *= inv; b.z *= inv; b.w *= inv;
    *(float4*)&p[tid * 4]        = a;
    *(float4*)&p[tid * 4 + 1024] = b;
}

// ---------------------------------------------------------------------------
extern "C" void kernel_launch(void* const* d_in, const int* in_sizes, int n_in,
                              void* d_out, int out_size, void* d_ws, size_t ws_size,
                              hipStream_t stream)
{
    const float* x    = (const float*)d_in[0];
    const int*   mask = (const int*)d_in[1];
    const float* Wq   = (const float*)d_in[2];
    const float* Wk   = (const float*)d_in[3];
    const float* Wv   = (const float*)d_in[4];
    const float* Wo   = (const float*)d_in[5];

    float* out  = (float*)d_out;                                   // (B,S,D)
    float* attn = out + (size_t)BATCH * SEQ * DMODEL;              // (B,H,S,S)

    // bf16 workspace: Q(BS x D), K(BS x KVDIM), V(BS x KVDIM), heads-out(BS x D)
    const size_t MROWS = (size_t)BATCH * SEQ;                      // 4096
    __bf16* Qws = (__bf16*)d_ws;
    __bf16* Kws = Qws + MROWS * DMODEL;
    __bf16* Vws = Kws + MROWS * KVDIM;
    __bf16* Hws = Vws + MROWS * KVDIM;
    (void)in_sizes; (void)n_in; (void)out_size; (void)ws_size;

    const dim3 blk(256);
    const long long SS = (long long)SEQ * SEQ;

    // 1) Q/K/V projections (fp32 x fp32 -> bf16 workspace)
    gemm_wmma_kernel<float, float, __bf16, false, false, false>
        <<<dim3(DMODEL / BN, MROWS / BM, 1), blk, 0, stream>>>(
            x, Wq, Qws, (int)MROWS, DMODEL, DMODEL, DMODEL, DMODEL, DMODEL,
            1, 1, 0, 0, 0, 0, 0, 0, 1.0f, nullptr);
    gemm_wmma_kernel<float, float, __bf16, false, false, false>
        <<<dim3(KVDIM / BN, MROWS / BM, 1), blk, 0, stream>>>(
            x, Wk, Kws, (int)MROWS, KVDIM, DMODEL, DMODEL, KVDIM, KVDIM,
            1, 1, 0, 0, 0, 0, 0, 0, 1.0f, nullptr);
    gemm_wmma_kernel<float, float, __bf16, false, false, false>
        <<<dim3(KVDIM / BN, MROWS / BM, 1), blk, 0, stream>>>(
            x, Wv, Vws, (int)MROWS, KVDIM, DMODEL, DMODEL, KVDIM, KVDIM,
            1, 1, 0, 0, 0, 0, 0, 0, 1.0f, nullptr);

    // 2) scores = scale * Q @ K^T, causal mask, per (b,h); written to attn
    gemm_wmma_kernel<__bf16, __bf16, float, true, true, false>
        <<<dim3(SEQ / BN, SEQ / BM, BATCH * NHEADS), blk, 0, stream>>>(
            Qws, Kws, attn, SEQ, SEQ, HEADDIM, DMODEL, KVDIM, SEQ,
            NHEADS, KVGROUPS,
            (long long)SEQ * DMODEL, HEADDIM,
            (long long)SEQ * KVDIM, HEADDIM,
            (long long)NHEADS * SS, SS,
            ATTNSCALE, mask);

    // 3) softmax rows in place
    softmax_rows_kernel<<<dim3(BATCH * NHEADS * SEQ), blk, 0, stream>>>(attn);

    // 4) heads-out = attn @ V (K loop truncated at the diagonal)
    gemm_wmma_kernel<float, __bf16, __bf16, false, false, true>
        <<<dim3(HEADDIM / BN, SEQ / BM, BATCH * NHEADS), blk, 0, stream>>>(
            attn, Vws, Hws, SEQ, HEADDIM, SEQ, SEQ, KVDIM, DMODEL,
            NHEADS, KVGROUPS,
            (long long)NHEADS * SS, SS,
            (long long)SEQ * KVDIM, HEADDIM,
            (long long)SEQ * DMODEL, HEADDIM,
            1.0f, nullptr);

    // 5) out = merged-heads @ Wo (fp32 result)
    gemm_wmma_kernel<__bf16, float, float, false, false, false>
        <<<dim3(DMODEL / BN, MROWS / BM, 1), blk, 0, stream>>>(
            Hws, Wo, out, (int)MROWS, DMODEL, DMODEL, DMODEL, DMODEL, DMODEL,
            1, 1, 0, 0, 0, 0, 0, 0, 1.0f, nullptr);
}
